// SparseAttention_83133386982045
// MI455X (gfx1250) — matrix operand order repaired and checked by reading, not compiled, not run on hardware
//
#include <hip/hip_runtime.h>
#include <math.h>

typedef _Float16 half_t;
typedef __attribute__((ext_vector_type(16))) _Float16 v16h;
typedef __attribute__((ext_vector_type(8)))  _Float16 v8h;
typedef __attribute__((ext_vector_type(8)))  float    v8f;

#define NEG_INF_F (-10000000000.0f)

constexpr int Bb = 2, Nn = 4096, Cc = 256, Hh = 8, HD = 32;
constexpr int M_ROWS = Bb * Nn;      // 8192
constexpr int QKV_N  = 3 * Cc;       // 768
constexpr float SCALE = 0.1767766952966369f; // 32^-0.5

// ---------- workspace layout (bytes) ----------
constexpr size_t OFF_XH     = 0;                                   // (8192,256) f16
constexpr size_t SZ_XH      = (size_t)M_ROWS * Cc * 2;
constexpr size_t OFF_WQKVT  = OFF_XH + SZ_XH;                      // (768,256)  f16 (transposed)
constexpr size_t SZ_WQKVT   = (size_t)QKV_N * Cc * 2;
constexpr size_t OFF_WPROJT = OFF_WQKVT + SZ_WQKVT;                // (256,256)  f16 (transposed)
constexpr size_t SZ_WPROJT  = (size_t)Cc * Cc * 2;
constexpr size_t OFF_Q      = OFF_WPROJT + SZ_WPROJT;              // (B,H,N,hd) f16, pre-scaled
constexpr size_t SZ_HEADS   = (size_t)Bb * Hh * Nn * HD * 2;
constexpr size_t OFF_K      = OFF_Q + SZ_HEADS;                    // (B,H,N,hd) f16
constexpr size_t OFF_VT     = OFF_K + SZ_HEADS;                    // (B,H,hd,N) f16 (transposed)
constexpr size_t OFF_AO     = OFF_VT + SZ_HEADS;                   // (B,N,C)    f16 attn output
constexpr size_t OFF_DEN    = OFF_AO + SZ_XH;                      // (8192)     f32
constexpr size_t OFF_KEYADD = OFF_DEN + (size_t)M_ROWS * 4;        // (8192)     f32 additive key mask

// ---------- small helpers ----------
__device__ inline v8f zero8() {
  v8f z;
#pragma unroll
  for (int i = 0; i < 8; ++i) z[i] = 0.0f;
  return z;
}
// A-fragment (16x32 f16): per lane, 16 contiguous halves at p
__device__ inline v16h ldA(const half_t* p) {
  v8h lo = *(const v8h*)p;
  v8h hi = *(const v8h*)(p + 8);
  return __builtin_shufflevector(lo, hi, 0,1,2,3,4,5,6,7,8,9,10,11,12,13,14,15);
}
// B-fragment (32x16 f16): per lane, 16 contiguous halves at p
__device__ inline v16h ldB(const half_t* p) {
  v8h lo = *(const v8h*)p;
  v8h hi = *(const v8h*)(p + 8);
  return __builtin_shufflevector(lo, hi, 0,1,2,3,4,5,6,7,8,9,10,11,12,13,14,15);
}
__device__ inline v8f wmma16(v16h a, v16h b, v8f c) {
  return __builtin_amdgcn_wmma_f32_16x16x32_f16(false, a, false, b, (short)0, c, false, false);
}

// ---------- K0: transpose + f16-convert weights ----------
__global__ void k_prep_w(const float* __restrict__ wqkv, const float* __restrict__ wproj,
                         half_t* __restrict__ wqkvT, half_t* __restrict__ wprojT) {
  int idx = blockIdx.x * blockDim.x + threadIdx.x;
  const int tot1 = Cc * QKV_N;                 // 196608
  if (idx < tot1) {
    int k = idx / QKV_N, nout = idx % QKV_N;   // wqkv is (in=256, out=768)
    wqkvT[(size_t)nout * Cc + k] = (half_t)wqkv[idx];
  } else {
    int idx2 = idx - tot1;
    if (idx2 < Cc * Cc) {
      int k = idx2 / Cc, nout = idx2 % Cc;
      wprojT[(size_t)nout * Cc + k] = (half_t)wproj[idx2];
    }
  }
}

// ---------- K1: density per row + x -> f16 ----------
__global__ void k_density(const float* __restrict__ x, half_t* __restrict__ xh,
                          float* __restrict__ density) {
  int wave = threadIdx.x >> 5;
  int lane = threadIdx.x & 31;
  int row  = blockIdx.x * 8 + wave;            // 0..8191, one wave per row
  int n = row & (Nn - 1);
  const float* xr = x + (size_t)row * Cc;
  float s = 0.f, s2 = 0.f, ad = 0.f;
  float vals[8];
#pragma unroll
  for (int e = 0; e < 8; ++e) {
    float v = xr[lane * 8 + e];
    vals[e] = v;
    s += v; s2 += v * v;
  }
  if (n > 0) {
    const float* xp = xr - Cc;
#pragma unroll
    for (int e = 0; e < 8; ++e) ad += fabsf(vals[e] - xp[lane * 8 + e]);
  }
#pragma unroll
  for (int e = 0; e < 8; ++e)
    xh[(size_t)row * Cc + lane * 8 + e] = (half_t)vals[e];
#pragma unroll
  for (int m = 16; m >= 1; m >>= 1) {
    s  += __shfl_xor(s, m, 32);
    s2 += __shfl_xor(s2, m, 32);
    ad += __shfl_xor(ad, m, 32);
  }
  if (lane == 0) {
    float mean = s / (float)Cc;
    float var  = (s2 - s * mean) / (float)(Cc - 1);   // ddof=1
    float stdv = sqrtf(var > 0.f ? var : 0.f);
    float grad = (n > 0) ? (ad / (float)Cc) : 0.f;
    density[row] = 0.5f * (stdv + grad);
  }
}

// ---------- K2: global threshold + per-key additive mask ----------
__global__ void k_threshold(const float* __restrict__ density, float* __restrict__ keyadd) {
  __shared__ float sh_s[32], sh_s2[32];
  __shared__ float sh_th;
  int tid = threadIdx.x;                       // 1024 threads, 1 block
  float s = 0.f, s2 = 0.f;
  for (int i = tid; i < M_ROWS; i += 1024) {
    float d = density[i];
    s += d; s2 += d * d;
  }
#pragma unroll
  for (int m = 16; m >= 1; m >>= 1) {
    s  += __shfl_xor(s, m, 32);
    s2 += __shfl_xor(s2, m, 32);
  }
  if ((tid & 31) == 0) { sh_s[tid >> 5] = s; sh_s2[tid >> 5] = s2; }
  __syncthreads();
  if (tid == 0) {
    float ts = 0.f, ts2 = 0.f;
    for (int i = 0; i < 32; ++i) { ts += sh_s[i]; ts2 += sh_s2[i]; }
    float mean = ts / (float)M_ROWS;
    float var  = (ts2 - ts * mean) / (float)(M_ROWS - 1);  // ddof=1
    sh_th = mean + 0.5f * sqrtf(var > 0.f ? var : 0.f);
  }
  __syncthreads();
  float th = sh_th;
  for (int i = tid; i < M_ROWS; i += 1024)
    keyadd[i] = (density[i] > th) ? 0.0f : NEG_INF_F;
}

// ---------- K3: QKV GEMM (WMMA), scatter into q / k / vT head layouts ----------
__global__ void k_qkv_gemm(const half_t* __restrict__ xh, const half_t* __restrict__ wqkvT,
                           half_t* __restrict__ qh, half_t* __restrict__ kh,
                           half_t* __restrict__ vT) {
  int wave = threadIdx.x >> 5, lane = threadIdx.x & 31;
  int job = blockIdx.x * 8 + wave;             // 6144 jobs total
  int mtile  = job / 12;                       // 512 M tiles of 16 rows
  int nchunk = job % 12;                       // 12 chunks of 64 output cols
  int hi16 = lane >> 4, ln = lane & 15;
  int row = mtile * 16 + ln;

  v8f acc[4];
#pragma unroll
  for (int t = 0; t < 4; ++t) acc[t] = zero8();

  const half_t* aBase = xh + (size_t)row * Cc + (hi16 ? 16 : 0);
  for (int k0 = 0; k0 < Cc; k0 += 32) {
    v16h a = ldA(aBase + k0);
#pragma unroll
    for (int t = 0; t < 4; ++t) {
      int col = nchunk * 64 + t * 16 + ln;
      v16h bf = ldB(wqkvT + (size_t)col * Cc + k0 + (hi16 ? 16 : 0));
      acc[t] = wmma16(a, bf, acc[t]);
    }
  }

  int m0 = hi16 * 8;
#pragma unroll
  for (int t = 0; t < 4; ++t) {
    int col0 = nchunk * 64 + t * 16;
    int s   = col0 >> 8;                       // 0=q, 1=k, 2=v
    int rem = col0 & 255;
    int h   = rem >> 5;
    int d   = (rem & 31) + ln;
#pragma unroll
    for (int r = 0; r < 8; ++r) {
      int rowg = mtile * 16 + m0 + r;
      int b = rowg >> 12;
      int nn = rowg & (Nn - 1);
      float v = acc[t][r];
      if (s == 0)
        qh[(((size_t)(b * Hh + h)) * Nn + nn) * HD + d] = (half_t)(v * SCALE);
      else if (s == 1)
        kh[(((size_t)(b * Hh + h)) * Nn + nn) * HD + d] = (half_t)v;
      else
        vT[(((size_t)(b * Hh + h)) * HD + d) * Nn + nn] = (half_t)v;
    }
  }
}

// ---------- K4: flash attention, one wave per (b,h,16-query tile) ----------
__global__ void __launch_bounds__(32) k_flash(const half_t* __restrict__ qh,
                                              const half_t* __restrict__ kh,
                                              const half_t* __restrict__ vT,
                                              const float* __restrict__ keyadd,
                                              half_t* __restrict__ ao) {
  __shared__ half_t P[16 * 32];                // P tile staging (D-layout -> A-layout)
  int lane = threadIdx.x & 31;
  int bid = blockIdx.x;
  int itile = bid & 255;
  int h = (bid >> 8) & 7;
  int b = bid >> 11;
  int i0 = itile * 16;
  int hi16 = lane >> 4, ln = lane & 15;
  const size_t headoff = ((size_t)(b * Hh + h)) * Nn * HD;

  v16h qa = ldA(qh + headoff + (size_t)(i0 + ln) * HD + (hi16 ? 16 : 0));

  v8f o0 = zero8(), o1 = zero8();
  float mrow[8], lrow[8];
#pragma unroll
  for (int r = 0; r < 8; ++r) { mrow[r] = -1e30f; lrow[r] = 0.f; }

  const float* ka = keyadd + b * Nn;
  const half_t* vb = vT + ((size_t)(b * Hh + h) * HD) * Nn;

  for (int j0 = 0; j0 < Nn; j0 += 32) {
    // prefetch next chunk's K rows and V columns into cache (global_prefetch_b8)
    if (j0 + 32 < Nn) {
      __builtin_prefetch(kh + headoff + (size_t)(j0 + 32 + ln) * HD + (hi16 ? 16 : 0), 0, 1);
      __builtin_prefetch(vb + (size_t)ln * Nn + j0 + 32 + (hi16 ? 16 : 0), 0, 1);
      __builtin_prefetch(vb + (size_t)(16 + ln) * Nn + j0 + 32 + (hi16 ? 16 : 0), 0, 1);
    }
    // S = Q K^T over 32 keys (two 16x16 tiles); q already carries the 1/sqrt(hd) scale
    v8f sv[2];
#pragma unroll
    for (int t = 0; t < 2; ++t) {
      v16h kb = ldB(kh + headoff + (size_t)(j0 + t * 16 + ln) * HD + (hi16 ? 16 : 0));
      sv[t] = wmma16(qa, kb, zero8());
    }
    // additive mask: local window OR high-info key
#pragma unroll
    for (int t = 0; t < 2; ++t) {
      int j = j0 + t * 16 + ln;
      float add = ka[j];
#pragma unroll
      for (int r = 0; r < 8; ++r) {
        int i = i0 + hi16 * 8 + r;
        int dd = i - j;
        sv[t][r] += (dd < 16 && dd > -16) ? 0.f : add;
      }
    }
    // online softmax update per query row
#pragma unroll
    for (int r = 0; r < 8; ++r) {
      float rm = fmaxf(sv[0][r], sv[1][r]);
#pragma unroll
      for (int msk = 1; msk <= 8; msk <<= 1) rm = fmaxf(rm, __shfl_xor(rm, msk, 32));
      float mnew  = fmaxf(mrow[r], rm);
      float alpha = __expf(mrow[r] - mnew);
      mrow[r] = mnew;
      float p0 = __expf(sv[0][r] - mnew);
      float p1 = __expf(sv[1][r] - mnew);
      int m = hi16 * 8 + r;
      P[m * 32 + ln]      = (half_t)p0;
      P[m * 32 + 16 + ln] = (half_t)p1;
      float ps = p0 + p1;
#pragma unroll
      for (int msk = 1; msk <= 8; msk <<= 1) ps += __shfl_xor(ps, msk, 32);
      lrow[r] = lrow[r] * alpha + ps;
      o0[r] *= alpha;
      o1[r] *= alpha;
    }
    __syncthreads();
    // O += P @ V   (P as A-fragment from LDS; V^T gives contiguous B-fragments)
    v16h pa = ldA(P + (size_t)ln * 32 + (hi16 ? 16 : 0));
    v16h v0 = ldB(vb + (size_t)(0  + ln) * Nn + j0 + (hi16 ? 16 : 0));
    v16h v1 = ldB(vb + (size_t)(16 + ln) * Nn + j0 + (hi16 ? 16 : 0));
    o0 = wmma16(pa, v0, o0);
    o1 = wmma16(pa, v1, o1);
    __syncthreads();
  }

  // normalize + store attn output (B,N,C) with col = h*32 + d
#pragma unroll
  for (int r = 0; r < 8; ++r) {
    int m = hi16 * 8 + r;
    float inv = 1.0f / lrow[r];
    size_t base = ((size_t)(b * Nn + i0 + m)) * Cc + h * HD;
    ao[base + ln]      = (half_t)(o0[r] * inv);
    ao[base + 16 + ln] = (half_t)(o1[r] * inv);
  }
}

// ---------- K5: output projection GEMM (WMMA) + bias, f32 out ----------
__global__ void k_proj(const half_t* __restrict__ ao, const half_t* __restrict__ wprojT,
                       const float* __restrict__ bias, float* __restrict__ out) {
  int wave = threadIdx.x >> 5, lane = threadIdx.x & 31;
  int job = blockIdx.x * 8 + wave;             // 2048 jobs
  int mtile  = job >> 2;
  int nchunk = job & 3;
  int hi16 = lane >> 4, ln = lane & 15;
  int row = mtile * 16 + ln;

  v8f acc[4];
#pragma unroll
  for (int t = 0; t < 4; ++t) acc[t] = zero8();

  const half_t* aBase = ao + (size_t)row * Cc + (hi16 ? 16 : 0);
  for (int k0 = 0; k0 < Cc; k0 += 32) {
    v16h a = ldA(aBase + k0);
#pragma unroll
    for (int t = 0; t < 4; ++t) {
      int col = nchunk * 64 + t * 16 + ln;
      v16h bf = ldB(wprojT + (size_t)col * Cc + k0 + (hi16 ? 16 : 0));
      acc[t] = wmma16(a, bf, acc[t]);
    }
  }

  int m0 = hi16 * 8;
#pragma unroll
  for (int t = 0; t < 4; ++t) {
    int col = nchunk * 64 + t * 16 + ln;
    float bv = bias[col];
#pragma unroll
    for (int r = 0; r < 8; ++r)
      out[(size_t)(mtile * 16 + m0 + r) * Cc + col] = acc[t][r] + bv;
  }
}

// ---------- launch ----------
extern "C" void kernel_launch(void* const* d_in, const int* in_sizes, int n_in,
                              void* d_out, int out_size, void* d_ws, size_t ws_size,
                              hipStream_t stream) {
  const float* x      = (const float*)d_in[0];
  const float* w_qkv  = (const float*)d_in[1];
  const float* w_proj = (const float*)d_in[2];
  const float* b_proj = (const float*)d_in[3];
  float* out = (float*)d_out;

  char* ws = (char*)d_ws;
  half_t* xh      = (half_t*)(ws + OFF_XH);
  half_t* wqkvT   = (half_t*)(ws + OFF_WQKVT);
  half_t* wprojT  = (half_t*)(ws + OFF_WPROJT);
  half_t* qh      = (half_t*)(ws + OFF_Q);
  half_t* kh      = (half_t*)(ws + OFF_K);
  half_t* vT      = (half_t*)(ws + OFF_VT);
  half_t* ao      = (half_t*)(ws + OFF_AO);
  float*  density = (float*)(ws + OFF_DEN);
  float*  keyadd  = (float*)(ws + OFF_KEYADD);

  // K0: weight transpose + f16 convert  (196608 + 65536 elements)
  k_prep_w<<<(Cc * QKV_N + Cc * Cc + 255) / 256, 256, 0, stream>>>(w_qkv, w_proj, wqkvT, wprojT);
  // K1: density + x->f16  (8 rows per 256-thread block)
  k_density<<<M_ROWS / 8, 256, 0, stream>>>(x, xh, density);
  // K2: threshold + key mask
  k_threshold<<<1, 1024, 0, stream>>>(density, keyadd);
  // K3: QKV GEMM  (512 mtiles * 12 nchunks = 6144 wave jobs, 8 waves/block)
  k_qkv_gemm<<<6144 / 8, 256, 0, stream>>>(xh, wqkvT, qh, kh, vT);
  // K4: flash attention  (B*H*(N/16) = 4096 blocks, 1 wave each)
  k_flash<<<Bb * Hh * (Nn / 16), 32, 0, stream>>>(qh, kh, vT, keyadd, ao);
  // K5: projection GEMM + bias  (512 mtiles * 4 nchunks = 2048 wave jobs)
  k_proj<<<2048 / 8, 256, 0, stream>>>(ao, wprojT, b_proj, out);
}
